// MiniDPF_84980222919170
// MI455X (gfx1250) — matrix operand order repaired and checked by reading, not compiled, not run on hardware
//
#include <hip/hip_runtime.h>

#define DEV static __device__ __forceinline__

typedef unsigned short u16;
typedef unsigned int   u32;

typedef __attribute__((ext_vector_type(8)))  __bf16 v8bf;
typedef __attribute__((ext_vector_type(16))) __bf16 v16bf;
typedef __attribute__((ext_vector_type(8)))  float  v8f;

// ---- problem constants ----
constexpr int BATCH = 2048;
constexpr int NPART = 1000;
constexpr int NROWS = 1008;   // 63 tiles of 16
constexpr int NTILE = 63;
constexpr int PP    = 72;     // particle row pitch (bf16 elems), 144B = 9*16B aligned
constexpr int HP    = 136;    // hidden row / WT2 pitch (bf16 elems), 272B = 17*16B
constexpr int NWAVE = 8;
constexpr int TOPK  = 32;

// ---- LDS layout (bytes), all regions 16B aligned ----
constexpr u32 OFF_PB    = 0;                          // particles bf16 1008x72
constexpr u32 OFF_WT1   = OFF_PB    + NROWS*PP*2;     // W_t1^T bf16 128x72 (K=64 used)
constexpr u32 OFF_WT2   = OFF_WT1   + 128*PP*2;       // W_t2^T bf16 64x136 (K=128)
constexpr u32 OFF_WW1   = OFF_WT2   + 64*HP*2;        // W_w1^T bf16 64x72 (K=64)
constexpr u32 OFF_SCR   = OFF_WW1   + 64*PP*2;        // per-wave scratch 8 x (16x136 bf16)
constexpr u32 OFF_LOGW  = OFF_SCR   + NWAVE*16*HP*2;  // 1008 f32
constexpr u32 OFF_SUMS  = OFF_LOGW  + NROWS*4;        // 16x64 f32 mean partials
constexpr u32 OFF_OBSIN = OFF_SUMS  + 16*64*4;        // 256 f32
constexpr u32 OFF_H1    = OFF_OBSIN + 256*4;          // 128 f32
constexpr u32 OFF_OBSE  = OFF_H1    + 128*4;          // 64 f32
constexpr u32 OFF_MEAN  = OFF_OBSE  + 64*4;           // 64 f32
constexpr u32 OFF_B1A   = OFF_MEAN  + 64*4;           // 128 f32 (b_t1 + regime fold)
constexpr u32 OFF_B2    = OFF_B1A   + 128*4;          // 64 f32
constexpr u32 OFF_B1W   = OFF_B2    + 64*4;           // 64 f32 (b_w1 + obs fold)
constexpr u32 OFF_W2    = OFF_B1W   + 64*4;           // 64 f32
constexpr u32 OFF_RP    = OFF_W2    + 64*4;           // 4 f32 (pad to 16)
constexpr u32 OFF_RED   = OFF_RP    + 16;             // 256 f32 reduction
constexpr u32 OFF_SELV  = OFF_RED   + 256*4;          // 32 f32
constexpr u32 OFF_SELI  = OFF_SELV  + 32*4;           // 32 i32
constexpr u32 LDS_BYTES = OFF_SELI  + 32*4;           // = 237,776 B < 320KB

// ---- helpers ----
DEV u16 f2bf(float x) {                // native f32 -> bf16 convert
    return __builtin_bit_cast(u16, (__bf16)x);
}
DEV float bf2f(u16 h) { return __uint_as_float(((u32)h) << 16); }

// exact GELU (one-time obs encoder only)
DEV float gelu_exact(float x) { return 0.5f * x * (1.0f + erff(x * 0.70710678118654752f)); }

// branch-free tanh-form GELU for the hot loop:
// gelu(x) ~= x * (1 - 1/(exp(2*0.79788456*(x + 0.044715 x^3)) + 1))
DEV float gelu_fast(float x) {
    float p  = x * x;
    float w  = __builtin_fmaf(0.0713548162726f, p, 1.5957691216057308f);
    float e  = __expf(x * w);
    float r  = __builtin_amdgcn_rcpf(e + 1.0f);
    return x - x * r;
}

DEV v8bf ld8(const u16* p) { return *reinterpret_cast<const v8bf*>(p); }
DEV v16bf cat16(v8bf a, v8bf b) {
    return __builtin_shufflevector(a, b, 0,1,2,3,4,5,6,7,8,9,10,11,12,13,14,15);
}
// A fragment 16x32 bf16 from a row-major row (ISA 7.12.2 16-bit A layout):
// lane(h=lane>>4, m=lane&15): elems 0..7 <- K=k0+8h.., elems 8..15 <- K=k0+16+8h..
DEV v16bf afrag(const u16* row, int k0, int h) {
    return cat16(ld8(row + k0 + h * 8), ld8(row + k0 + 16 + h * 8));
}
// B fragment 32x16 bf16 from K-contiguous column storage, same K order as A:
// elems 0..7 <- K=k0+8h.., elems 8..15 <- K=k0+16+8h..
// col points at W^T[n][*] (length >= k0+32).
DEV v16bf bfrag(const u16* col, int k0, int h) {
    return cat16(ld8(col + k0 + h * 8), ld8(col + k0 + 16 + h * 8));
}
DEV v8f wmma_bf16(v16bf a, v16bf b, v8f c) {
    return __builtin_amdgcn_wmma_f32_16x16x32_bf16(false, a, false, b, (short)0, c, false, false);
}
DEV v8f splat8(float x) {
    v8f r;
#pragma unroll
    for (int i = 0; i < 8; i++) r[i] = x;
    return r;
}

__global__ void __launch_bounds__(256, 1)
dpf_fused_kernel(const float* __restrict__ obs,  const float* __restrict__ pin,
                 const float* __restrict__ Wo1,  const float* __restrict__ bo1,
                 const float* __restrict__ Wo2,  const float* __restrict__ bo2,
                 const float* __restrict__ Wt1,  const float* __restrict__ bt1,
                 const float* __restrict__ Wt2,  const float* __restrict__ bt2,
                 const float* __restrict__ Ww1,  const float* __restrict__ bw1,
                 const float* __restrict__ Ww2,  const float* __restrict__ bw2,
                 const float* __restrict__ Wr,   const float* __restrict__ br,
                 float* __restrict__ out)
{
    extern __shared__ char smem[];
    u16*   PB   = (u16*)(smem + OFF_PB);
    u16*   WT1  = (u16*)(smem + OFF_WT1);
    u16*   WT2  = (u16*)(smem + OFF_WT2);
    u16*   WW1  = (u16*)(smem + OFF_WW1);
    float* LOGW = (float*)(smem + OFF_LOGW);
    float* SUMS = (float*)(smem + OFF_SUMS);
    float* OBSI = (float*)(smem + OFF_OBSIN);
    float* H1   = (float*)(smem + OFF_H1);
    float* OBSE = (float*)(smem + OFF_OBSE);
    float* MEAN = (float*)(smem + OFF_MEAN);
    float* B1A  = (float*)(smem + OFF_B1A);
    float* B2   = (float*)(smem + OFF_B2);
    float* B1W  = (float*)(smem + OFF_B1W);
    float* W2   = (float*)(smem + OFF_W2);
    float* RP   = (float*)(smem + OFF_RP);
    float* RED  = (float*)(smem + OFF_RED);
    float* SELV = (float*)(smem + OFF_SELV);
    int*   SELI = (int*)(smem + OFF_SELI);

    const int tid = threadIdx.x;
    const int b   = blockIdx.x;
    const int lane = tid & 31, wid = tid >> 5;
    const int h = lane >> 4, nn = lane & 15, mrow = lane & 15;

    // ================= Phase 1: stage particles (x0.1 -> bf16 LDS) + mean partials =================
    {
        const float4* pin4 = reinterpret_cast<const float4*>(pin + (size_t)b * NPART * 64);
        float s0 = 0.f, s1 = 0.f, s2 = 0.f, s3 = 0.f;
        for (int e = tid; e < NPART * 16; e += 256) {   // 16 float4 per particle row
            float4 v = pin4[e];
            float x0 = v.x * 0.1f, x1 = v.y * 0.1f, x2 = v.z * 0.1f, x3 = v.w * 0.1f;
            int rowi = e >> 4, d0 = (e & 15) << 2;
            u32 lo = (u32)f2bf(x0) | ((u32)f2bf(x1) << 16);
            u32 hi = (u32)f2bf(x2) | ((u32)f2bf(x3) << 16);
            u32* dst = (u32*)(PB + rowi * PP + d0);
            dst[0] = lo; dst[1] = hi;
            s0 += x0; s1 += x1; s2 += x2; s3 += x3;
        }
        int d0 = (tid & 15) << 2, g = tid >> 4;
        SUMS[g * 64 + d0 + 0] = s0; SUMS[g * 64 + d0 + 1] = s1;
        SUMS[g * 64 + d0 + 2] = s2; SUMS[g * 64 + d0 + 3] = s3;
        OBSI[tid] = obs[(size_t)b * 256 + tid];
        if (tid < 8) LOGW[NPART + tid] = -3.0e38f;      // pad rows
    }
    __syncthreads();

    // ====== Phase 2: mean, obs hidden, weight transposes (coalesced f4 loads + ds scatter) ======
    {   // W_t1^T: 64 latent rows x 128 cols, row-major float4 chunks
        const float4* W4 = reinterpret_cast<const float4*>(Wt1);
        for (int idx = tid; idx < 64 * 32; idx += 256) {     // 2048 float4
            int k = idx >> 5, n0 = (idx & 31) << 2;
            float4 v = W4[idx];
            WT1[(n0 + 0) * PP + k] = f2bf(v.x);
            WT1[(n0 + 1) * PP + k] = f2bf(v.y);
            WT1[(n0 + 2) * PP + k] = f2bf(v.z);
            WT1[(n0 + 3) * PP + k] = f2bf(v.w);
        }
    }
    {   // W_t2^T: 128 rows x 64 cols
        const float4* W4 = reinterpret_cast<const float4*>(Wt2);
        for (int idx = tid; idx < 128 * 16; idx += 256) {    // 2048 float4
            int k = idx >> 4, n0 = (idx & 15) << 2;
            float4 v = W4[idx];
            WT2[(n0 + 0) * HP + k] = f2bf(v.x);
            WT2[(n0 + 1) * HP + k] = f2bf(v.y);
            WT2[(n0 + 2) * HP + k] = f2bf(v.z);
            WT2[(n0 + 3) * HP + k] = f2bf(v.w);
        }
    }
    {   // W_w1^T: first 64 rows x 64 cols
        const float4* W4 = reinterpret_cast<const float4*>(Ww1);
        for (int idx = tid; idx < 64 * 16; idx += 256) {     // 1024 float4
            int k = idx >> 4, n0 = (idx & 15) << 2;
            float4 v = W4[idx];
            WW1[(n0 + 0) * PP + k] = f2bf(v.x);
            WW1[(n0 + 1) * PP + k] = f2bf(v.y);
            WW1[(n0 + 2) * PP + k] = f2bf(v.z);
            WW1[(n0 + 3) * PP + k] = f2bf(v.w);
        }
    }
    if (tid < 128) {                                    // obs encoder layer 1
        float a = bo1[tid];
        for (int k = 0; k < 256; k++) a += OBSI[k] * Wo1[k * 128 + tid];
        H1[tid] = gelu_exact(a);
    }
    if (tid < 64) {                                     // particle mean (already x0.1)
        float m = 0.f;
        for (int g = 0; g < 16; g++) m += SUMS[g * 64 + tid];
        MEAN[tid] = m * (1.0f / NPART);
    }
    __syncthreads();

    // ================= Phase 3: obs encoding, regime softmax =================
    if (tid < 64) {
        float a = bo2[tid];
        for (int k = 0; k < 128; k++) a += H1[k] * Wo2[k * 64 + tid];
        OBSE[tid] = a;
    }
    if (tid == 0) {
        float lg[4], mx = -3.0e38f;
        for (int j = 0; j < 4; j++) {
            float a = br[j];
            for (int d = 0; d < 64; d++) a += MEAN[d] * Wr[d * 4 + j];
            lg[j] = a; mx = fmaxf(mx, a);
        }
        float s = 0.f;
        for (int j = 0; j < 4; j++) { lg[j] = __expf(lg[j] - mx); s += lg[j]; }
        for (int j = 0; j < 4; j++) {
            float p = lg[j] / s;
            RP[j] = p;
            out[(size_t)BATCH * 64 + (size_t)b * 4 + j] = p;   // regime_probs output
        }
    }
    __syncthreads();

    // ================= Phase 4: fold concat-constants into biases =================
    if (tid < 128) {    // b_t1 + W_t1[64..67,:] . regime_probs
        float a = bt1[tid];
        for (int j = 0; j < 4; j++) a += RP[j] * Wt1[(64 + j) * 128 + tid];
        B1A[tid] = a;
    }
    if (tid < 64) {     // b_w1 + W_w1[64..127,:] . obs_encoded ; copy b_t2, W_w2
        B2[tid] = bt2[tid];
        float a = bw1[tid];
        for (int d = 0; d < 64; d++) a += OBSE[d] * Ww1[(64 + d) * 64 + tid];
        B1W[tid] = a;
        W2[tid] = Ww2[tid];
    }
    __syncthreads();

    // ================= Phase 5: main WMMA tile loop (per-wave, 16 rows/tile) =================
    {
        u16*   scr  = (u16*)(smem + OFF_SCR) + wid * 16 * HP;     // wave-private 16x128 bf16
        float* scrF = (float*)(smem + OFF_SCR) + wid * 1088;      // same bytes as 16x68 f32
        const float bw2s = bw2[0];

        for (int tile = wid; tile < NTILE; tile += NWAVE) {
            const int row0 = tile * 16;
            // --- transition layer 1: hidden(16x128) = P(16x64) @ Wt1 + foldbias ---
            const u16* arow = PB + (row0 + mrow) * PP;
            v16bf a0 = afrag(arow, 0, h);
            v16bf a1 = afrag(arow, 32, h);
#pragma unroll
            for (int t = 0; t < 8; t++) {
                v8f acc = splat8(B1A[t * 16 + nn]);
                const u16* bcol = WT1 + (t * 16 + nn) * PP;
                acc = wmma_bf16(a0, bfrag(bcol, 0, h), acc);
                acc = wmma_bf16(a1, bfrag(bcol, 32, h), acc);
#pragma unroll
                for (int r = 0; r < 8; r++)          // D layout: m = r + 8h, col = t*16+nn
                    scr[(r + 8 * h) * HP + t * 16 + nn] = f2bf(gelu_fast(acc[r]));
            }
            // --- transition layer 2: p_new(16x64) = hidden @ Wt2 + b_t2 (in-place to PB) ---
            const u16* ar2 = scr + mrow * HP;
            v16bf a20 = afrag(ar2, 0, h);
            v16bf a21 = afrag(ar2, 32, h);
            v16bf a22 = afrag(ar2, 64, h);
            v16bf a23 = afrag(ar2, 96, h);
#pragma unroll
            for (int t = 0; t < 4; t++) {
                v8f acc = splat8(B2[t * 16 + nn]);
                const u16* bcol = WT2 + (t * 16 + nn) * HP;
                acc = wmma_bf16(a20, bfrag(bcol, 0, h), acc);
                acc = wmma_bf16(a21, bfrag(bcol, 32, h), acc);
                acc = wmma_bf16(a22, bfrag(bcol, 64, h), acc);
                acc = wmma_bf16(a23, bfrag(bcol, 96, h), acc);
#pragma unroll
                for (int r = 0; r < 8; r++)
                    PB[(row0 + r + 8 * h) * PP + t * 16 + nn] = f2bf(acc[r]);
            }
            // --- weight net layer 1: h2(16x64) = p_new @ Ww1 + foldbias, GELU -> f32 scratch ---
            const u16* ar3 = PB + (row0 + mrow) * PP;
            v16bf a30 = afrag(ar3, 0, h);
            v16bf a31 = afrag(ar3, 32, h);
#pragma unroll
            for (int t = 0; t < 4; t++) {
                v8f acc = splat8(B1W[t * 16 + nn]);
                const u16* bcol = WW1 + (t * 16 + nn) * PP;
                acc = wmma_bf16(a30, bfrag(bcol, 0, h), acc);
                acc = wmma_bf16(a31, bfrag(bcol, 32, h), acc);
#pragma unroll
                for (int r = 0; r < 8; r++)
                    scrF[(r + 8 * h) * 68 + t * 16 + nn] = gelu_fast(acc[r]);
            }
            // --- log weights: dot(h2_row, w2) + b_w2 ---
            if (lane < 16) {
                int rg = row0 + lane;
                if (rg < NPART) {
                    float s = bw2s;
                    const float4* hr = reinterpret_cast<const float4*>(scrF + lane * 68);
                    const float4* w4 = reinterpret_cast<const float4*>(W2);
                    for (int j = 0; j < 16; j++) {
                        float4 a = hr[j], w = w4[j];
                        s += a.x * w.x + a.y * w.y + a.z * w.z + a.w * w.w;
                    }
                    LOGW[rg] = s;
                }
            }
        }
    }
    __syncthreads();

    // ================= Phase 6: softmax over N =================
    {
        float lmax = -3.0e38f;
        for (int i = tid; i < NPART; i += 256) lmax = fmaxf(lmax, LOGW[i]);
        RED[tid] = lmax;
        __syncthreads();
        for (int s = 128; s > 0; s >>= 1) {
            if (tid < s) RED[tid] = fmaxf(RED[tid], RED[tid + s]);
            __syncthreads();
        }
        float mx = RED[0];
        __syncthreads();
        float lsum = 0.f;
        for (int i = tid; i < NPART; i += 256) {
            float e = __expf(LOGW[i] - mx);
            LOGW[i] = e;                      // LOGW now holds unnormalized weights
            lsum += e;
        }
        RED[tid] = lsum;
        __syncthreads();
        for (int s = 128; s > 0; s >>= 1) {
            if (tid < s) RED[tid] += RED[tid + s];
            __syncthreads();
        }
        // RED[0] = denom, preserved through top-k
    }

    // ================= Phase 7: top-32 (wave 0, shuffle reductions, no barriers) ==========
    if (wid == 0) {
        for (int it = 0; it < TOPK; it++) {
            float bv = -2.0f; int bi = 0;
            for (int i = lane; i < NPART; i += 32) {
                float v = LOGW[i];
                if (v > bv) { bv = v; bi = i; }
            }
#pragma unroll
            for (int off = 16; off >= 1; off >>= 1) {
                float ov = __shfl_xor(bv, off);
                int   oi = __shfl_xor(bi, off);
                if (ov > bv || (ov == bv && oi < bi)) { bv = ov; bi = oi; }
            }
            if (lane == 0) {
                SELV[it] = bv; SELI[it] = bi;
                LOGW[bi] = -1.0f;             // exclude (weights are >= 0)
            }
        }
    }
    __syncthreads();

    // ================= Phase 8: weighted gather of top particles =================
    if (tid < 64) {
        float inv = 1.0f / RED[0];
        float acc = 0.f;
        for (int k = 0; k < TOPK; k++)
            acc += (SELV[k] * inv) * bf2f(PB[SELI[k] * PP + tid]);
        out[(size_t)b * 64 + tid] = acc;
    }
}

extern "C" void kernel_launch(void* const* d_in, const int* in_sizes, int n_in,
                              void* d_out, int out_size, void* d_ws, size_t ws_size,
                              hipStream_t stream) {
    (void)in_sizes; (void)n_in; (void)d_ws; (void)ws_size; (void)out_size;
    const float* obs = (const float*)d_in[0];
    const float* pin = (const float*)d_in[1];
    const float* Wo1 = (const float*)d_in[2];
    const float* bo1 = (const float*)d_in[3];
    const float* Wo2 = (const float*)d_in[4];
    const float* bo2 = (const float*)d_in[5];
    const float* Wt1 = (const float*)d_in[6];
    const float* bt1 = (const float*)d_in[7];
    const float* Wt2 = (const float*)d_in[8];
    const float* bt2 = (const float*)d_in[9];
    const float* Ww1 = (const float*)d_in[10];
    const float* bw1 = (const float*)d_in[11];
    const float* Ww2 = (const float*)d_in[12];
    const float* bw2 = (const float*)d_in[13];
    const float* Wr  = (const float*)d_in[14];
    const float* br  = (const float*)d_in[15];
    float* out = (float*)d_out;

    dpf_fused_kernel<<<BATCH, 256, LDS_BYTES, stream>>>(
        obs, pin, Wo1, bo1, Wo2, bo2, Wt1, bt1, Wt2, bt2,
        Ww1, bw1, Ww2, bw2, Wr, br, out);
}
